// HydraEncoderLayer_37366215475341
// MI455X (gfx1250) — matrix operand order repaired and checked by reading, not compiled, not run on hardware
//
#include <hip/hip_runtime.h>
#include <hip/hip_bf16.h>

// ---------------------------------------------------------------------------
// Hydra encoder layer for gfx1250 (MI455X), bf16 WMMA path.
// B=8, S=4096, D=512, DI=2048, M=B*S=32768.
// ---------------------------------------------------------------------------

#define B_DIM  8
#define S_DIM  4096
#define D_DIM  512
#define DI_DIM 2048
#define M_DIM  (B_DIM * S_DIM)   // 32768
#define EPSN   1e-12f

typedef __attribute__((ext_vector_type(16))) __bf16 v16bf;
typedef __attribute__((ext_vector_type(8)))  float  v8f;
typedef __attribute__((ext_vector_type(4)))  int    i4v;

// address-space qualified int4 pointers for the async-to-LDS builtin
typedef __attribute__((address_space(1))) i4v* gas_i4p;  // global
typedef __attribute__((address_space(3))) i4v* las_i4p;  // LDS

union FragAB {
    uint4 u[2];
    v16bf v;
};

// epilogue flags
#define FL_SUMSQ 1   // atomicAdd per-row sum of squares of raw GEMM output
#define FL_BIAS  2   // add bias[n]
#define FL_RELU  4   // relu
#define FL_RES   8   // add f32 residual [M,N]
#define FL_STF32 16  // store f32 result
#define FL_STBF  32  // store bf16 result

#if defined(__has_builtin)
#if __has_builtin(__builtin_amdgcn_global_load_async_to_lds_b128)
#define USE_ASYNC_LDS 1
#endif
#endif
#ifndef USE_ASYNC_LDS
#define USE_ASYNC_LDS 0
#endif

// ---------------------------------------------------------------------------
// Tiled bf16 GEMM:  C[M,N] = A[M,K] * B[K,N],  Bt given as [N][K] row-major.
// Block tile 128(M) x 128(N), K-step 32, ping-pong LDS double buffering.
// 256 threads = 8 wave32 waves arranged 4(M) x 2(N); each wave owns a
// 32x64 sub-tile = 2 A-frags x 4 B-frags = 8 v_wmma_f32_16x16x32_bf16 per
// K-step (12 ds_load_b128 per 8 WMMAs).
// M, N, K are multiples of 128/128/32 for every call here (no edge handling).
// ---------------------------------------------------------------------------
template <int FLAGS>
__global__ __launch_bounds__(256) void gemm_bf16_kernel(
    const __bf16* __restrict__ A,   // [M,K]
    const __bf16* __restrict__ Bt,  // [N,K]
    int M, int N, int K,
    __bf16* __restrict__ outB,      // [M,N] bf16 (FL_STBF)
    float*  __restrict__ outF,      // [M,N] f32  (FL_STF32)
    const float* __restrict__ bias, // [N]        (FL_BIAS)
    const float* __restrict__ res,  // [M,N] f32  (FL_RES)
    float* __restrict__ sumsq)      // [M]        (FL_SUMSQ)
{
    __shared__ __align__(16) __bf16 As[2][128][40];  // 40 = 32 + pad (80B rows)
    __shared__ __align__(16) __bf16 Bs[2][128][40];

    const int tid  = threadIdx.x;
    const int wave = tid >> 5;        // 0..7
    const int lane = tid & 31;        // 0..31
    const int mg   = wave >> 1;       // 0..3 : 32-row group
    const int ng   = wave & 1;        // 0..1 : 64-col group
    const int m0 = blockIdx.y * 128;
    const int n0 = blockIdx.x * 128;

    v8f c[2][4];
#pragma unroll
    for (int i = 0; i < 2; ++i)
#pragma unroll
        for (int j = 0; j < 4; ++j) c[i][j] = (v8f)0.0f;

    // global->LDS staging assignment: 128 rows x 32 cols per tile,
    // thread t handles row t/2, 16-element half (t&1) -> 2 x b128 each.
    const int srow = tid >> 1;            // 0..127
    const int scol = (tid & 1) << 4;      // 0 or 16 (bf16 elems)
    const __bf16* ag = A  + (size_t)(m0 + srow) * K + scol;
    const __bf16* bg = Bt + (size_t)(n0 + srow) * K + scol;

    const int ra    = (mg << 5) + (lane & 15);    // A row 0..127 (frag 0)
    const int khalf = lane >> 4;                  // 0 or 1
    const int cbb   = (ng << 6) + (lane & 15);    // B col base

    const int niter = K >> 5;

#if USE_ASYNC_LDS
#define STAGE(p, kk)                                                         \
    do {                                                                     \
        __builtin_amdgcn_global_load_async_to_lds_b128(                      \
            (gas_i4p)(ag + (kk)),     (las_i4p)&As[p][srow][scol],     0, 0);\
        __builtin_amdgcn_global_load_async_to_lds_b128(                      \
            (gas_i4p)(ag + (kk) + 8), (las_i4p)&As[p][srow][scol + 8], 0, 0);\
        __builtin_amdgcn_global_load_async_to_lds_b128(                      \
            (gas_i4p)(bg + (kk)),     (las_i4p)&Bs[p][srow][scol],     0, 0);\
        __builtin_amdgcn_global_load_async_to_lds_b128(                      \
            (gas_i4p)(bg + (kk) + 8), (las_i4p)&Bs[p][srow][scol + 8], 0, 0);\
    } while (0)

    STAGE(0, 0);
    asm volatile("s_wait_asynccnt 0x0" ::: "memory");
    __syncthreads();
#else
    {
        uint4 sa0 = *(const uint4*)(ag);
        uint4 sa1 = *(const uint4*)(ag + 8);
        uint4 sb0 = *(const uint4*)(bg);
        uint4 sb1 = *(const uint4*)(bg + 8);
        *(uint4*)&As[0][srow][scol]     = sa0;
        *(uint4*)&As[0][srow][scol + 8] = sa1;
        *(uint4*)&Bs[0][srow][scol]     = sb0;
        *(uint4*)&Bs[0][srow][scol + 8] = sb1;
        __syncthreads();
    }
#endif

    for (int it = 0; it < niter; ++it) {
        const int p = it & 1;
        const bool more = (it + 1) < niter;

#if USE_ASYNC_LDS
        if (more) STAGE(p ^ 1, (size_t)(it + 1) << 5);
#else
        uint4 sa0, sa1, sb0, sb1;
        if (more) {
            const size_t kk = (size_t)(it + 1) << 5;
            sa0 = *(const uint4*)(ag + kk);
            sa1 = *(const uint4*)(ag + kk + 8);
            sb0 = *(const uint4*)(bg + kk);
            sb1 = *(const uint4*)(bg + kk + 8);
        }
#endif
        // ---- compute on buffer p ----
        // A fragment layout: lane<16 -> K {0..7,16..23}; lane>=16 -> {8..15,24..31}
        FragAB a0, a1;
        a0.u[0] = *(const uint4*)&As[p][ra][khalf * 8];
        a0.u[1] = *(const uint4*)&As[p][ra][16 + khalf * 8];
        a1.u[0] = *(const uint4*)&As[p][ra + 16][khalf * 8];
        a1.u[1] = *(const uint4*)&As[p][ra + 16][16 + khalf * 8];

#pragma unroll
        for (int j = 0; j < 4; ++j) {
            // B fragment: lane<16 -> col lane, K {0..7,16..23}; lane>=16 -> K {8..15,24..31}
            const int cb = cbb + (j << 4);
            FragAB b;
            b.u[0] = *(const uint4*)&Bs[p][cb][khalf * 8];
            b.u[1] = *(const uint4*)&Bs[p][cb][16 + khalf * 8];
            c[0][j] = __builtin_amdgcn_wmma_f32_16x16x32_bf16(
                          false, a0.v, false, b.v, (short)0, c[0][j], false, false);
            c[1][j] = __builtin_amdgcn_wmma_f32_16x16x32_bf16(
                          false, a1.v, false, b.v, (short)0, c[1][j], false, false);
        }

#if USE_ASYNC_LDS
        if (more) asm volatile("s_wait_asynccnt 0x0" ::: "memory");
        __syncthreads();
#else
        if (more) {
            const int q = p ^ 1;
            *(uint4*)&As[q][srow][scol]     = sa0;
            *(uint4*)&As[q][srow][scol + 8] = sa1;
            *(uint4*)&Bs[q][srow][scol]     = sb0;
            *(uint4*)&Bs[q][srow][scol + 8] = sb1;
        }
        __syncthreads();
#endif
    }

    // ---- epilogue ------------------------------------------------------
    const int colb    = lane & 15;
    const int rowhalf = (lane >> 4) << 3;  // 0 or 8

#pragma unroll
    for (int i16 = 0; i16 < 2; ++i16) {
        if (FLAGS & FL_SUMSQ) {
#pragma unroll
            for (int ii = 0; ii < 8; ++ii) {
                float ss = 0.0f;
#pragma unroll
                for (int j = 0; j < 4; ++j) { float t = c[i16][j][ii]; ss += t * t; }
                // reduce over the 16 lanes that share this row
                ss += __shfl_xor(ss, 1, 32);
                ss += __shfl_xor(ss, 2, 32);
                ss += __shfl_xor(ss, 4, 32);
                ss += __shfl_xor(ss, 8, 32);
                if ((lane & 15) == 0) {
                    const int m = m0 + (mg << 5) + (i16 << 4) + rowhalf + ii;
                    atomicAdd(&sumsq[m], ss);
                }
            }
        }
#pragma unroll
        for (int j = 0; j < 4; ++j) {
#pragma unroll
            for (int ii = 0; ii < 8; ++ii) {
                const int m = m0 + (mg << 5) + (i16 << 4) + rowhalf + ii;
                const int n = n0 + (ng << 6) + (j << 4) + colb;
                float v = c[i16][j][ii];
                if (FLAGS & FL_BIAS) v += bias[n];
                if (FLAGS & FL_RELU) v = v > 0.0f ? v : 0.0f;
                if (FLAGS & FL_RES)  v += res[(size_t)m * N + n];
                if (FLAGS & FL_STF32) outF[(size_t)m * N + n] = v;
                if (FLAGS & FL_STBF)  outB[(size_t)m * N + n] = (__bf16)v;
            }
        }
    }
}

// ---------------------------------------------------------------------------
// elementwise helpers
// ---------------------------------------------------------------------------
__global__ void zero_f32_kernel(float* __restrict__ p, int n) {
    int i = blockIdx.x * blockDim.x + threadIdx.x;
    if (i < n) p[i] = 0.0f;
}

__global__ void cvt_bf16_kernel(const float* __restrict__ in,
                                __bf16* __restrict__ out, int n) {
    int i = blockIdx.x * blockDim.x + threadIdx.x;
    if (i < n) out[i] = (__bf16)in[i];
}

// out[n*K + k] = (bf16) in[k*N + n]   (weights -> [N][K] bf16)
__global__ void cvt_t_bf16_kernel(const float* __restrict__ in,
                                  __bf16* __restrict__ out, int K, int N) {
    int i = blockIdx.x * blockDim.x + threadIdx.x;
    if (i < K * N) {
        int n = i / K, k = i - n * K;
        out[i] = (__bf16)in[(size_t)k * N + n];
    }
}

// kv[b,e] = sum_s  k[b,s,e]/max(||k_row||,eps) * v[b,s,e]
__global__ void kv_reduce_kernel(const __bf16* __restrict__ Kb,
                                 const __bf16* __restrict__ Vb,
                                 const float* __restrict__ ssk,
                                 float* __restrict__ kv) {
    const int e = blockIdx.x * blockDim.x + threadIdx.x;  // 0..DI-1
    const int b = blockIdx.y;
    const __bf16* kp = Kb + (size_t)b * S_DIM * DI_DIM + e;
    const __bf16* vp = Vb + (size_t)b * S_DIM * DI_DIM + e;
    const float*  sp = ssk + (size_t)b * S_DIM;
    float acc = 0.0f;
    for (int s = 0; s < S_DIM; ++s) {
        const float inv = 1.0f / fmaxf(sqrtf(sp[s]), EPSN);
        acc += (float)kp[(size_t)s * DI_DIM] * inv * (float)vp[(size_t)s * DI_DIM];
    }
    kv[(size_t)b * DI_DIM + e] = acc;
}

// Q[b,s,e] *= kv[b,e] / max(||q_row||, eps)   (in place, bf16)
__global__ void scale_q_kernel(__bf16* __restrict__ Qb,
                               const float* __restrict__ ssq,
                               const float* __restrict__ kv) {
    const size_t i = (size_t)blockIdx.x * blockDim.x + threadIdx.x;
    const int    e   = (int)(i % DI_DIM);
    const size_t row = i / DI_DIM;          // b*S + s
    const int    b   = (int)(row / S_DIM);
    const float inv = 1.0f / fmaxf(sqrtf(ssq[row]), EPSN);
    Qb[i] = (__bf16)((float)Qb[i] * inv * kv[(size_t)b * DI_DIM + e]);
}

// ---------------------------------------------------------------------------
// launch
// ---------------------------------------------------------------------------
extern "C" void kernel_launch(void* const* d_in, const int* in_sizes, int n_in,
                              void* d_out, int out_size, void* d_ws, size_t ws_size,
                              hipStream_t stream) {
    (void)in_sizes; (void)n_in; (void)out_size; (void)ws_size;

    const float* x  = (const float*)d_in[0];  // [M, D]
    const float* wq = (const float*)d_in[1];  // [D, DI]
    const float* wk = (const float*)d_in[2];
    const float* wv = (const float*)d_in[3];
    const float* wo = (const float*)d_in[4];  // [DI, D]
    const float* w1 = (const float*)d_in[5];  // [D, DI]
    const float* b1 = (const float*)d_in[6];  // [DI]
    const float* w2 = (const float*)d_in[7];  // [DI, D]
    const float* b2 = (const float*)d_in[8];  // [D]
    float* out = (float*)d_out;               // [M, D]

    // workspace carve-up (256B aligned)
    size_t off = 0;
    auto alloc = [&](size_t bytes) {
        void* p = (char*)d_ws + off;
        off += (bytes + 255) & ~(size_t)255;
        return p;
    };
    __bf16* xb    = (__bf16*)alloc((size_t)M_DIM * D_DIM * 2);
    __bf16* wqT   = (__bf16*)alloc((size_t)DI_DIM * D_DIM * 2);
    __bf16* wkT   = (__bf16*)alloc((size_t)DI_DIM * D_DIM * 2);
    __bf16* wvT   = (__bf16*)alloc((size_t)DI_DIM * D_DIM * 2);
    __bf16* woT   = (__bf16*)alloc((size_t)D_DIM * DI_DIM * 2);
    __bf16* w1T   = (__bf16*)alloc((size_t)DI_DIM * D_DIM * 2);
    __bf16* w2T   = (__bf16*)alloc((size_t)D_DIM * DI_DIM * 2);
    __bf16* Qb    = (__bf16*)alloc((size_t)M_DIM * DI_DIM * 2);
    __bf16* Kb    = (__bf16*)alloc((size_t)M_DIM * DI_DIM * 2);
    __bf16* Vb    = (__bf16*)alloc((size_t)M_DIM * DI_DIM * 2);
    float*  ssq   = (float*)alloc((size_t)M_DIM * 4);
    float*  ssk   = (float*)alloc((size_t)M_DIM * 4);
    float*  kv    = (float*)alloc((size_t)B_DIM * DI_DIM * 4);
    float*  attnF = (float*)alloc((size_t)M_DIM * D_DIM * 4);
    __bf16* attnB = (__bf16*)alloc((size_t)M_DIM * D_DIM * 2);
    __bf16* Hb    = Kb;  // K consumed by kv_reduce before H is produced

    // 1) precision conversion + weight transposition
    {
        int n = M_DIM * D_DIM;
        cvt_bf16_kernel<<<(n + 255) / 256, 256, 0, stream>>>(x, xb, n);
        int w = D_DIM * DI_DIM;
        int g = (w + 255) / 256;
        cvt_t_bf16_kernel<<<g, 256, 0, stream>>>(wq, wqT, D_DIM, DI_DIM);
        cvt_t_bf16_kernel<<<g, 256, 0, stream>>>(wk, wkT, D_DIM, DI_DIM);
        cvt_t_bf16_kernel<<<g, 256, 0, stream>>>(wv, wvT, D_DIM, DI_DIM);
        cvt_t_bf16_kernel<<<g, 256, 0, stream>>>(wo, woT, DI_DIM, D_DIM);
        cvt_t_bf16_kernel<<<g, 256, 0, stream>>>(w1, w1T, D_DIM, DI_DIM);
        cvt_t_bf16_kernel<<<g, 256, 0, stream>>>(w2, w2T, DI_DIM, D_DIM);
        zero_f32_kernel<<<(M_DIM + 255) / 256, 256, 0, stream>>>(ssq, M_DIM);
        zero_f32_kernel<<<(M_DIM + 255) / 256, 256, 0, stream>>>(ssk, M_DIM);
    }

    const dim3 blk(256);
    const dim3 gDI(DI_DIM / 128, M_DIM / 128);  // N=2048 GEMMs
    const dim3 gD (D_DIM  / 128, M_DIM / 128);  // N=512  GEMMs

    // 2) Q,K,V projections (with row sum-of-squares for Q,K)
    gemm_bf16_kernel<FL_SUMSQ | FL_STBF><<<gDI, blk, 0, stream>>>(
        xb, wqT, M_DIM, DI_DIM, D_DIM, Qb, nullptr, nullptr, nullptr, ssq);
    gemm_bf16_kernel<FL_SUMSQ | FL_STBF><<<gDI, blk, 0, stream>>>(
        xb, wkT, M_DIM, DI_DIM, D_DIM, Kb, nullptr, nullptr, nullptr, ssk);
    gemm_bf16_kernel<FL_STBF><<<gDI, blk, 0, stream>>>(
        xb, wvT, M_DIM, DI_DIM, D_DIM, Vb, nullptr, nullptr, nullptr, nullptr);

    // 3) kv = sum_s normalize(k)*v ;  q <- normalize(q)*kv
    kv_reduce_kernel<<<dim3(DI_DIM / 256, B_DIM), blk, 0, stream>>>(Kb, Vb, ssk, kv);
    {
        size_t n = (size_t)M_DIM * DI_DIM;
        scale_q_kernel<<<(unsigned)(n / 256), blk, 0, stream>>>(Qb, ssq, kv);
    }

    // 4) attn = (q*kv) @ wo + x
    gemm_bf16_kernel<FL_RES | FL_STF32 | FL_STBF><<<gD, blk, 0, stream>>>(
        Qb, woT, M_DIM, D_DIM, DI_DIM, attnB, attnF, nullptr, x, nullptr);

    // 5) h = relu(attn @ w1 + b1)
    gemm_bf16_kernel<FL_BIAS | FL_RELU | FL_STBF><<<gDI, blk, 0, stream>>>(
        attnB, w1T, M_DIM, DI_DIM, D_DIM, Hb, nullptr, b1, nullptr, nullptr);

    // 6) out = h @ w2 + b2 + attn
    gemm_bf16_kernel<FL_BIAS | FL_RES | FL_STF32><<<gD, blk, 0, stream>>>(
        Hb, w2T, M_DIM, D_DIM, DI_DIM, nullptr, out, b2, attnF, nullptr);
}
